// GCN_JKNet_48206712930326
// MI455X (gfx1250) — matrix operand order repaired and checked by reading, not compiled, not run on hardware
//
#include <hip/hip_runtime.h>
#include <hip/hip_bf16.h>

// Problem constants (match reference)
#define N_NODES 100000
#define F_IN_   512
#define E_EDGES 3200000
#define C_HID_  16
#define H_      32
#define C_OUT_  40

typedef __attribute__((ext_vector_type(16))) __bf16 v16bf;
typedef __attribute__((ext_vector_type(8)))  float  v8f;

__device__ __forceinline__ float sigmoidf_(float v) {
    return 1.0f / (1.0f + __expf(-v));
}

__device__ __forceinline__ float wave_sum(float v) {
#pragma unroll
    for (int off = 16; off > 0; off >>= 1) v += __shfl_xor(v, off, 32);
    return v;
}

// ---------------- degree / norm ----------------
__global__ void k_init_deg(float* deg) {
    int i = blockIdx.x * blockDim.x + threadIdx.x;
    if (i < N_NODES) deg[i] = 1.0f;  // self-loop
}

__global__ void k_deg_accum(const int* __restrict__ col, float* deg) {
    int e = blockIdx.x * blockDim.x + threadIdx.x;
    if (e < E_EDGES) atomicAdd(&deg[col[e]], 1.0f);
}

__global__ void k_dinv(const float* __restrict__ deg, float* dinv) {
    int i = blockIdx.x * blockDim.x + threadIdx.x;
    if (i < N_NODES) {
        float d = deg[i];
        dinv[i] = (d > 0.0f) ? __frsqrt_rn(d) : 0.0f;
    }
}

// ---------------- stage W1 into per-lane WMMA B-fragment order (bf16) ----------------
// st[((kb*32 + lane)<<4) + j] = bf16( W1[(kb*32 + (lane>>4)*8 + (j&7) + (j>>3)*16) * 16 + (lane&15)] )
// so each lane's 16-element B fragment for K-block kb is one contiguous 32B vector.
__global__ void k_stage_w1(const float* __restrict__ w1, __bf16* __restrict__ st) {
    int t = blockIdx.x * blockDim.x + threadIdx.x;
    if (t >= (F_IN_ / 32) * 32 * 16) return;
    int j    = t & 15;
    int lane = (t >> 4) & 31;
    int kb   = t >> 9;
    int k    = kb * 32 + (lane >> 4) * 8 + (j & 7) + (j >> 3) * 16;
    int col  = lane & 15;
    st[t] = (__bf16)w1[k * C_HID_ + col];
}

// ---------------- stage W2 (16x16) as one zero-padded K=32 B fragment ----------------
// j<8 carry K=khalf*8+j (real W2), j>=8 carry K=16+khalf*8+(j-8) (zeros).
__global__ void k_stage_w2(const float* __restrict__ w2, __bf16* __restrict__ st) {
    int t = blockIdx.x * blockDim.x + threadIdx.x;
    if (t >= 32 * 16) return;
    int j = t & 15, lane = t >> 4;
    int khalf = lane >> 4, col = lane & 15;
    st[t] = (j < 8) ? (__bf16)w2[(khalf * 8 + j) * C_HID_ + col] : (__bf16)0.0f;
}

// ---------------- big GEMM: xw = x[N,512] @ W1[512,16] via WMMA bf16 ----------------
// One wave32 per 16-row tile; fully unrolled K loop: 16 x v_wmma_f32_16x16x32_bf16.
__global__ __launch_bounds__(256) void k_gemm_xw1(const float* __restrict__ x,
                                                  const __bf16* __restrict__ w1st,
                                                  float* __restrict__ xw) {
    int wave = (blockIdx.x * blockDim.x + threadIdx.x) >> 5;
    int lane = threadIdx.x & 31;
    int m0 = wave * 16;
    if (m0 >= N_NODES) return;  // wave-uniform: EXEC stays all-ones for WMMA

    const int mrow  = lane & 15;
    const int khalf = lane >> 4;
    const float4* xrow =
        reinterpret_cast<const float4*>(x + (size_t)(m0 + mrow) * F_IN_ + khalf * 8);

    v8f acc = {};
#pragma unroll
    for (int kb = 0; kb < F_IN_ / 32; ++kb) {
        // A fragment (16x32 bf16): a[0..7] = K kb*32+khalf*8..+7, a[8..15] = +16
        float4 p0 = xrow[kb * 8 + 0];
        float4 p1 = xrow[kb * 8 + 1];
        float4 p2 = xrow[kb * 8 + 4];
        float4 p3 = xrow[kb * 8 + 5];
        v16bf a;
        a[0]  = (__bf16)p0.x; a[1]  = (__bf16)p0.y; a[2]  = (__bf16)p0.z; a[3]  = (__bf16)p0.w;
        a[4]  = (__bf16)p1.x; a[5]  = (__bf16)p1.y; a[6]  = (__bf16)p1.z; a[7]  = (__bf16)p1.w;
        a[8]  = (__bf16)p2.x; a[9]  = (__bf16)p2.y; a[10] = (__bf16)p2.z; a[11] = (__bf16)p2.w;
        a[12] = (__bf16)p3.x; a[13] = (__bf16)p3.y; a[14] = (__bf16)p3.z; a[15] = (__bf16)p3.w;
        // B fragment: one contiguous 32B per-lane vector (pre-staged)
        v16bf b = *reinterpret_cast<const v16bf*>(w1st + (((size_t)kb * 32 + lane) << 4));
        acc = __builtin_amdgcn_wmma_f32_16x16x32_bf16(false, a, false, b,
                                                      (short)0, acc, false, false);
    }

    // C layout: lanes 0-15 -> M 0..7 in acc[0..7], lanes 16-31 -> M 8..15; N = lane&15
    const int n  = lane & 15;
    const int mb = khalf * 8;
    float* o = xw + (size_t)(m0 + mb) * C_HID_ + n;
#pragma unroll
    for (int r = 0; r < 8; ++r) o[(size_t)r * C_HID_] = acc[r];
}

// ---------------- small GEMM: o = a[N,16] @ W2[16,16] via single WMMA (K padded) ----
__global__ __launch_bounds__(256) void k_gemm16_wmma(const float* __restrict__ a,
                                                     const __bf16* __restrict__ w2st,
                                                     float* __restrict__ o) {
    int wave = (blockIdx.x * blockDim.x + threadIdx.x) >> 5;
    int lane = threadIdx.x & 31;
    int m0 = wave * 16;
    if (m0 >= N_NODES) return;
    const int mrow  = lane & 15;
    const int khalf = lane >> 4;
    // A: a[0..7] = K khalf*8..+7 (real), a[8..15] = K 16+khalf*8.. (zero pad)
    const float4* ar =
        reinterpret_cast<const float4*>(a + (size_t)(m0 + mrow) * C_HID_ + khalf * 8);
    float4 p0 = ar[0];
    float4 p1 = ar[1];
    v16bf af;
    af[0] = (__bf16)p0.x; af[1] = (__bf16)p0.y; af[2] = (__bf16)p0.z; af[3] = (__bf16)p0.w;
    af[4] = (__bf16)p1.x; af[5] = (__bf16)p1.y; af[6] = (__bf16)p1.z; af[7] = (__bf16)p1.w;
#pragma unroll
    for (int j = 8; j < 16; ++j) af[j] = (__bf16)0.0f;
    v16bf bf = *reinterpret_cast<const v16bf*>(w2st + ((size_t)lane << 4));
    v8f acc = {};
    acc = __builtin_amdgcn_wmma_f32_16x16x32_bf16(false, af, false, bf,
                                                  (short)0, acc, false, false);
    const int n  = lane & 15;
    const int mb = khalf * 8;
    float* op = o + (size_t)(m0 + mb) * C_HID_ + n;
#pragma unroll
    for (int r = 0; r < 8; ++r) op[(size_t)r * C_HID_] = acc[r];
}

// ---------------- zero ----------------
__global__ void k_zero(float* p, int n) {
    int t = blockIdx.x * blockDim.x + threadIdx.x;
    if (t < n) p[t] = 0.0f;
}

// ---------------- edge scatter: dst[col] += dinv[row]*dinv[col] * src[row] ----------------
// 16 lanes per edge (one per channel) -> coalesced 64B row traffic.
__global__ void k_scatter(const float* __restrict__ src, float* dst,
                          const int* __restrict__ row, const int* __restrict__ col,
                          const float* __restrict__ dinv) {
    unsigned t = blockIdx.x * blockDim.x + threadIdx.x;
    int e  = (int)(t >> 4);
    int ch = (int)(t & 15);
    if (e >= E_EDGES) return;
    int r = row[e], c = col[e];
    float nm = dinv[r] * dinv[c];
    atomicAdd(dst + (size_t)c * C_HID_ + ch, nm * src[(size_t)r * C_HID_ + ch]);
}

// ---------------- self-loop + bias + optional relu ----------------
__global__ void k_finish(const float* __restrict__ agg, const float* __restrict__ pre,
                         const float* __restrict__ dinv, const float* __restrict__ bias,
                         float* __restrict__ out, int relu) {
    int t = blockIdx.x * blockDim.x + threadIdx.x;
    if (t >= N_NODES * C_HID_) return;
    int i = t >> 4, c = t & 15;
    float d = dinv[i];
    float v = agg[t] + d * d * pre[t] + (bias ? bias[c] : 0.0f);
    out[t] = relu ? fmaxf(v, 0.0f) : v;
}

// ---------------- one LSTM step, one node per wave, lane = hidden channel ----------------
__device__ __forceinline__ void lstm_step_w(int lane, const float* __restrict__ xt,
                                            const float* __restrict__ wih,
                                            const float* __restrict__ whh,
                                            const float* __restrict__ bih,
                                            const float* __restrict__ bhh,
                                            float& h, float& c) {
    float gi = bih[lane]      + bhh[lane];
    float gf = bih[32 + lane] + bhh[32 + lane];
    float gg = bih[64 + lane] + bhh[64 + lane];
    float go = bih[96 + lane] + bhh[96 + lane];
#pragma unroll
    for (int k = 0; k < C_HID_; ++k) {
        float xv = xt[k];  // wave-uniform broadcast load
        gi += xv * wih[lane * C_HID_ + k];
        gf += xv * wih[(32 + lane) * C_HID_ + k];
        gg += xv * wih[(64 + lane) * C_HID_ + k];
        go += xv * wih[(96 + lane) * C_HID_ + k];
    }
    float hold = h;
#pragma unroll
    for (int k = 0; k < H_; ++k) {
        float hv = __shfl(hold, k, 32);  // cross-lane recurrence (ds_bpermute)
        gi += hv * whh[lane * H_ + k];
        gf += hv * whh[(32 + lane) * H_ + k];
        gg += hv * whh[(64 + lane) * H_ + k];
        go += hv * whh[(96 + lane) * H_ + k];
    }
    c = sigmoidf_(gf) * c + sigmoidf_(gi) * tanhf(gg);
    h = sigmoidf_(go) * tanhf(c);
}

// ---------------- bi-LSTM JumpingKnowledge + attention combine ----------------
__global__ __launch_bounds__(256) void k_lstm_jk(
    const float* __restrict__ x1, const float* __restrict__ x2,
    const float* __restrict__ wihf, const float* __restrict__ whhf,
    const float* __restrict__ bihf, const float* __restrict__ bhhf,
    const float* __restrict__ wihb, const float* __restrict__ whhb,
    const float* __restrict__ bihb, const float* __restrict__ bhhb,
    const float* __restrict__ attw, const float* __restrict__ attb,
    float* __restrict__ xjk) {
    unsigned gt = blockIdx.x * blockDim.x + threadIdx.x;
    int node = (int)(gt >> 5);
    int lane = (int)(gt & 31);
    if (node >= N_NODES) return;
    const float* xa = x1 + (size_t)node * C_HID_;
    const float* xb = x2 + (size_t)node * C_HID_;
    float s0 = 0.0f, s1 = 0.0f;
    {   // forward over [x1, x2]
        float h = 0.0f, c = 0.0f;
        lstm_step_w(lane, xa, wihf, whhf, bihf, bhhf, h, c);
        s0 += h * attw[lane];
        lstm_step_w(lane, xb, wihf, whhf, bihf, bhhf, h, c);
        s1 += h * attw[lane];
    }
    {   // backward over [x2, x1], outputs reversed: step0->t=1, step1->t=0
        float h = 0.0f, c = 0.0f;
        lstm_step_w(lane, xb, wihb, whhb, bihb, bhhb, h, c);
        s1 += h * attw[H_ + lane];
        lstm_step_w(lane, xa, wihb, whhb, bihb, bhhb, h, c);
        s0 += h * attw[H_ + lane];
    }
    s0 = wave_sum(s0) + attb[0];
    s1 = wave_sum(s1) + attb[0];
    float m  = fmaxf(s0, s1);
    float e0 = __expf(s0 - m), e1 = __expf(s1 - m);
    float inv = 1.0f / (e0 + e1);
    float a0 = e0 * inv, a1 = e1 * inv;
    if (lane < C_HID_)
        xjk[(size_t)node * C_HID_ + lane] = a0 * xa[lane] + a1 * xb[lane];
}

// ---------------- final linear 16->40 + log_softmax ----------------
__global__ void k_out(const float* __restrict__ xp, const float* __restrict__ lw,
                      const float* __restrict__ lb, float* __restrict__ out) {
    int i = blockIdx.x * blockDim.x + threadIdx.x;
    if (i >= N_NODES) return;
    float xr[C_HID_];
#pragma unroll
    for (int k = 0; k < C_HID_; ++k) xr[k] = xp[(size_t)i * C_HID_ + k];
    float o[C_OUT_];
    float m = -1e30f;
#pragma unroll
    for (int c = 0; c < C_OUT_; ++c) {
        float s = lb[c];
#pragma unroll
        for (int k = 0; k < C_HID_; ++k) s += xr[k] * lw[k * C_OUT_ + c];
        o[c] = s;
        m = fmaxf(m, s);
    }
    float lse = 0.0f;
#pragma unroll
    for (int c = 0; c < C_OUT_; ++c) lse += __expf(o[c] - m);
    lse = __logf(lse);
#pragma unroll
    for (int c = 0; c < C_OUT_; ++c) out[(size_t)i * C_OUT_ + c] = o[c] - m - lse;
}

extern "C" void kernel_launch(void* const* d_in, const int* in_sizes, int n_in,
                              void* d_out, int out_size, void* d_ws, size_t ws_size,
                              hipStream_t stream) {
    const float* x    = (const float*)d_in[0];
    const int*   ei   = (const int*)d_in[1];
    const float* W1   = (const float*)d_in[2];
    const float* b1   = (const float*)d_in[3];
    const float* W2   = (const float*)d_in[4];
    const float* b2   = (const float*)d_in[5];
    const float* wihf = (const float*)d_in[6];
    const float* whhf = (const float*)d_in[7];
    const float* bihf = (const float*)d_in[8];
    const float* bhhf = (const float*)d_in[9];
    const float* wihb = (const float*)d_in[10];
    const float* whhb = (const float*)d_in[11];
    const float* bihb = (const float*)d_in[12];
    const float* bhhb = (const float*)d_in[13];
    const float* attw = (const float*)d_in[14];
    const float* attb = (const float*)d_in[15];
    const float* linw = (const float*)d_in[16];
    const float* linb = (const float*)d_in[17];

    const int* row = ei;             // edge_index[0]
    const int* col = ei + E_EDGES;   // edge_index[1]

    // workspace carve-up (floats)
    float* ws   = (float*)d_ws;
    float* deg  = ws;                              // N
    float* dinv = deg + N_NODES;                   // N
    float* xw   = dinv + N_NODES;                  // N*16
    float* agg  = xw + (size_t)N_NODES * C_HID_;   // N*16
    float* x1   = agg + (size_t)N_NODES * C_HID_;  // N*16
    float* buf  = x1 + (size_t)N_NODES * C_HID_;   // N*16 (x1@W2, later xp)
    float* x2   = buf + (size_t)N_NODES * C_HID_;  // N*16
    float* xjk  = x2 + (size_t)N_NODES * C_HID_;   // N*16
    __bf16* w1st = (__bf16*)(xjk + (size_t)N_NODES * C_HID_);  // 8192 bf16 (staged)
    __bf16* w2st = w1st + (F_IN_ / 32) * 32 * 16;              // 512 bf16 (staged)

    const int TB = 256;
    const int gN   = (N_NODES + TB - 1) / TB;
    const int gNC  = (N_NODES * C_HID_ + TB - 1) / TB;
    const int gE   = (E_EDGES + TB - 1) / TB;
    const int gE16 = (int)(((long long)E_EDGES * 16 + TB - 1) / TB);
    const int gGEMM = ((N_NODES + 15) / 16 * 32 + TB - 1) / TB;  // one wave per 16 rows
    const int gLSTM = (N_NODES * 32 + TB - 1) / TB;              // one wave per node

    // normalization
    k_init_deg<<<gN, TB, 0, stream>>>(deg);
    k_deg_accum<<<gE, TB, 0, stream>>>(col, deg);
    k_dinv<<<gN, TB, 0, stream>>>(deg, dinv);

    // stage weights into WMMA fragment order
    k_stage_w1<<<((F_IN_ / 32) * 32 * 16 + TB - 1) / TB, TB, 0, stream>>>(W1, w1st);
    k_stage_w2<<<(32 * 16 + TB - 1) / TB, TB, 0, stream>>>(W2, w2st);

    // layer 1: WMMA GEMM, propagate, relu
    k_gemm_xw1<<<gGEMM, TB, 0, stream>>>(x, w1st, xw);
    k_zero<<<gNC, TB, 0, stream>>>(agg, N_NODES * C_HID_);
    k_scatter<<<gE16, TB, 0, stream>>>(xw, agg, row, col, dinv);
    k_finish<<<gNC, TB, 0, stream>>>(agg, xw, dinv, b1, x1, 1);

    // layer 2: WMMA (K=16 padded to 32), propagate, relu
    k_gemm16_wmma<<<gGEMM, TB, 0, stream>>>(x1, w2st, buf);
    k_zero<<<gNC, TB, 0, stream>>>(agg, N_NODES * C_HID_);
    k_scatter<<<gE16, TB, 0, stream>>>(buf, agg, row, col, dinv);
    k_finish<<<gNC, TB, 0, stream>>>(agg, buf, dinv, b2, x2, 1);

    // JumpingKnowledge bi-LSTM + attention
    k_lstm_jk<<<gLSTM, TB, 0, stream>>>(x1, x2, wihf, whhf, bihf, bhhf,
                                        wihb, whhb, bihb, bhhb, attw, attb, xjk);

    // APPNP K=1, alpha=0: one propagation (no bias, no relu) -> buf
    k_zero<<<gNC, TB, 0, stream>>>(agg, N_NODES * C_HID_);
    k_scatter<<<gE16, TB, 0, stream>>>(xjk, agg, row, col, dinv);
    k_finish<<<gNC, TB, 0, stream>>>(agg, xjk, dinv, nullptr, buf, 0);

    // final linear + log_softmax
    k_out<<<gN, TB, 0, stream>>>(buf, linw, linb, (float*)d_out);
}
